// LSTMClassifier_18296560681222
// MI455X (gfx1250) — matrix operand order repaired and checked by reading, compiled, not run on hardware
//
#include <hip/hip_runtime.h>
#include <hip/hip_bf16.h>

typedef __attribute__((ext_vector_type(16))) __bf16 v16bf;
typedef __attribute__((ext_vector_type(8)))  float  v8f;

#define S_LEN 256
#define BATCH 128
#define EDIM  300
#define HDIM  512
#define NGATE 2048              // 4*H
#define NT_ALL 128              // NGATE/16
#define KT_H  16                // H/32
#define KT_X0 10                // ceil(300/32)
#define KT_X1 32                // 1024/32
#define MT_ALL 2048             // S*B/16

__device__ __forceinline__ float sig_(float x) { return 1.0f / (1.0f + __expf(-x)); }

// tanh via native exp, safe for large |x|: t=exp(-2|x|) in (0,1]; result sign-restored
__device__ __forceinline__ float tanh_(float x) {
    float ax = __builtin_fabsf(x);
    float t  = __expf(-2.0f * ax);
    float r  = (1.0f - t) / (1.0f + t);
    return __builtin_copysignf(r, x);
}

#define WMMA_BF16(acc, a, b) \
  __builtin_amdgcn_wmma_f32_16x16x32_bf16(false, (a), false, (b), (short)0, (acc), false, false)

// ---------------- small setup kernels ----------------

__global__ void len_kernel(const float* __restrict__ x, int* __restrict__ len) {
    int b = threadIdx.x;
    if (b < BATCH) {
        int L = (int)x[((size_t)b * S_LEN + (S_LEN - 1)) * EDIM];
        len[b] = L < S_LEN ? L : S_LEN;
    }
}

// fill 3 h-state packed buffers with bf16(1.0), zero barrier counters
__global__ void init_kernel(__bf16* h0, __bf16* h1, __bf16* h2, unsigned* cnt) {
    int i = blockIdx.x * blockDim.x + threadIdx.x;
    if (i < 8 * KT_H * 512) {           // 65536 bf16 per buffer
        __bf16 one = (__bf16)1.0f;
        h0[i] = one; h1[i] = one; h2[i] = one;
    }
    if (i < 4) cnt[i] = 0u;
}

// Pack W [N=2048][Kreal] f32 row-major -> WMMA B-operand bf16 tiles
// layout: [nt][kt][lane][e], B[k][n] = W[n][k]; lane: n=nt*16+(lane&15),
// element e: k = kt*32 + ((e>>3)<<4) + (lane>>4)*8 + (e&7)
__global__ void pack_w_kernel(const float* __restrict__ W, __bf16* __restrict__ out,
                              int Kreal, int Ktiles) {
    size_t idx = (size_t)blockIdx.x * blockDim.x + threadIdx.x;
    size_t total = (size_t)NT_ALL * Ktiles * 512;
    if (idx >= total) return;
    int e    = idx & 15;
    int lane = (idx >> 4) & 31;
    int kt   = (int)((idx >> 9) % Ktiles);
    int nt   = (int)((idx >> 9) / Ktiles);
    int n = nt * 16 + (lane & 15);
    int k = kt * 32 + ((e >> 3) << 4) + (lane >> 4) * 8 + (e & 7);
    float v = (k < Kreal) ? W[(size_t)n * Kreal + k] : 0.0f;
    out[idx] = (__bf16)v;
}

// Pack x [B][S][E] -> A-operand bf16 tiles, logical row = t*128 + b, K padded 300->320
// A layout element e of lane: k = kt*32 + ((e>>3)<<4) + (lane>>4)*8 + (e&7); m = mt*16+(lane&15)
__global__ void pack_x_kernel(const float* __restrict__ x, __bf16* __restrict__ out) {
    size_t idx = (size_t)blockIdx.x * blockDim.x + threadIdx.x;
    size_t total = (size_t)MT_ALL * KT_X0 * 512;
    if (idx >= total) return;
    int e    = idx & 15;
    int lane = (idx >> 4) & 31;
    int kt   = (int)((idx >> 9) % KT_X0);
    int mt   = (int)((idx >> 9) / KT_X0);
    int m = mt * 16 + (lane & 15);
    int t = m >> 7, b = m & 127;
    int k = kt * 32 + ((e >> 3) << 4) + (lane >> 4) * 8 + (e & 7);
    float v = (k < EDIM) ? x[((size_t)b * S_LEN + t) * EDIM + k] : 0.0f;
    out[idx] = (__bf16)v;
}

// ---------------- big input-projection GEMM (bias folded in) ----------------
// C(pre) layout: [t][nt(128)][m8(8)][lane(32)*8r] f32 so the recurrent kernel
// loads its D-layout tile as one contiguous v8f per lane.
__global__ __launch_bounds__(256)
void gemm_pre_kernel(const __bf16* __restrict__ Ap, const __bf16* __restrict__ Bp,
                     const float* __restrict__ bias_a, const float* __restrict__ bias_b,
                     float* __restrict__ C, int Ktiles) {
    const int lane = threadIdx.x;          // 0..31
    const int w    = threadIdx.y;          // 0..7
    const int mt   = blockIdx.y * 8 + w;
    const int nb   = blockIdx.x * 8;
    const __bf16* arow = Ap + (size_t)mt * Ktiles * 512;

    int nlo = lane & 15;
    v8f z[8];
#pragma unroll
    for (int q = 0; q < 8; ++q) {
        float bz = bias_a[(nb + q) * 16 + nlo] + bias_b[(nb + q) * 16 + nlo];
#pragma unroll
        for (int r = 0; r < 8; ++r) z[q][r] = bz;
    }
    for (int kt = 0; kt < Ktiles; ++kt) {
        v16bf a = *(const v16bf*)(arow + (size_t)kt * 512 + lane * 16);
#pragma unroll
        for (int q = 0; q < 8; ++q) {
            v16bf b = *(const v16bf*)(Bp + (((size_t)(nb + q) * Ktiles + kt) * 32 + lane) * 16);
            z[q] = WMMA_BF16(z[q], a, b);
        }
    }
#pragma unroll
    for (int q = 0; q < 8; ++q) {
        size_t off = ((((size_t)(mt >> 3)) * NT_ALL + (nb + q)) * 8 + (mt & 7)) * 256 + lane * 8;
        *(v8f*)(C + off) = z[q];
    }
}

// ---------------- recurrent LSTM kernel (grid-synced, persistent state) ----------------
// grid (16, ndir); block (32,8). Workgroup g owns h columns [g*32, g*32+32).
// Wave w owns batch rows [w*16, w*16+16) (== A/m-tile w). c,h state lives in VGPRs.
__global__ __launch_bounds__(256)
void lstm_recur_kernel(const float* pre0, const float* pre1,
                       const __bf16* whh0, const __bf16* whh1,
                       __bf16* hpk0, __bf16* hpk1,
                       const int* __restrict__ lengths,
                       __bf16* a1dst,              // nullable: packed concat output (layer 0)
                       int koff0, int koff1,       // concat k-offset per dir
                       int rev0, int rev1,
                       unsigned* cnt,              // one counter per dir
                       float* hT)                  // nullable: final hidden f32 (layer 1)
{
    const int d = blockIdx.y;
    const float*  pre = d ? pre1 : pre0;
    const __bf16* whh = d ? whh1 : whh0;
    __bf16*       hpk = d ? hpk1 : hpk0;
    const int koff = d ? koff1 : koff0;
    const int rev  = d ? rev1  : rev0;
    unsigned* my_cnt = cnt + d;

    const int lane = threadIdx.x;      // 0..31
    const int w    = threadIdx.y;      // 0..7  (m-tile)
    const int g    = blockIdx.x;       // 0..15 (h-column slice, also hpk k-tile)
    const int half = lane >> 4;
    const int n15  = lane & 15;
    const int mrow = w * 16 + half * 8;

    int lenr[8];
#pragma unroll
    for (int r = 0; r < 8; ++r) lenr[r] = lengths[mrow + r];

    v8f hst[2], cst[2];
#pragma unroll
    for (int p = 0; p < 2; ++p)
#pragma unroll
        for (int r = 0; r < 8; ++r) { hst[p][r] = 1.0f; cst[p][r] = 1.0f; }

    for (int s = 0; s < S_LEN; ++s) {
        const int t = rev ? (S_LEN - 1 - s) : s;

        // z = pre[t] + h @ Whh^T  for this wg's 8 gate-column tiles
        v8f z[8];
#pragma unroll
        for (int q = 0; q < 8; ++q) {
            int nt = (q >> 1) * 32 + g * 2 + (q & 1);
            z[q] = *(const v8f*)(pre + (((size_t)t * NT_ALL + nt) * 8 + w) * 256 + lane * 8);
        }
        for (int kt = 0; kt < KT_H; ++kt) {
            v16bf a = *(const v16bf*)(hpk + (((size_t)w * KT_H + kt) * 32 + lane) * 16);
#pragma unroll
            for (int q = 0; q < 8; ++q) {
                int nt = (q >> 1) * 32 + g * 2 + (q & 1);
                v16bf b = *(const v16bf*)(whh + (((size_t)nt * KT_H + kt) * 32 + lane) * 16);
                z[q] = WMMA_BF16(z[q], a, b);
            }
        }

        // gates -> new state (masked), write next h in packed A layout
#pragma unroll
        for (int p = 0; p < 2; ++p) {
#pragma unroll
            for (int r = 0; r < 8; ++r) {
                float zi = z[0 + p][r], zf = z[2 + p][r], zg = z[4 + p][r], zo = z[6 + p][r];
                float cn = sig_(zf) * cst[p][r] + sig_(zi) * tanh_(zg);
                float hn = sig_(zo) * tanh_(cn);
                bool msk = t < lenr[r];
                float c2 = msk ? cn : cst[p][r];
                float h2 = msk ? hn : hst[p][r];
                cst[p][r] = c2; hst[p][r] = h2;

                int j = g * 32 + p * 16 + n15;           // h column
                int mA = mrow + r;                        // batch row
                // h-state buffer (Ktiles = KT_H), k = j
                {
                    int kk = j & 31;                      // ktile == g
                    int e  = ((kk >> 4) << 3) + (kk & 7);
                    int lA = (((kk >> 3) & 1) << 4) + (mA & 15);
                    hpk[(((size_t)w * KT_H + g) * 32 + lA) * 16 + e] = (__bf16)h2;
                }
                // layer-1 input (concat) buffer (Ktiles = KT_X1), row = t*128 + mA
                if (a1dst) {
                    int k1 = koff + j;
                    int kt1 = k1 >> 5, kk = k1 & 31;
                    int e  = ((kk >> 4) << 3) + (kk & 7);
                    int lA = (((kk >> 3) & 1) << 4) + (mA & 15);
                    size_t mt1 = (size_t)t * 8 + w;
                    a1dst[((mt1 * KT_X1 + kt1) * 32 + lA) * 16 + e] = (__bf16)h2;
                }
            }
        }

        // device-scope grid barrier for this direction (monotonic counter)
        __threadfence();
        __syncthreads();
        if (threadIdx.x == 0 && threadIdx.y == 0) {
            atomicAdd(my_cnt, 1u);
            unsigned target = gridDim.x * (unsigned)(s + 1);
            while (__hip_atomic_load(my_cnt, __ATOMIC_ACQUIRE, __HIP_MEMORY_SCOPE_AGENT) < target) {}
        }
        __syncthreads();
        __threadfence();
    }

    if (hT) {
#pragma unroll
        for (int p = 0; p < 2; ++p)
#pragma unroll
            for (int r = 0; r < 8; ++r)
                hT[(size_t)(mrow + r) * HDIM + g * 32 + p * 16 + n15] = hst[p][r];
    }
}

// ---------------- tiny FC head: out = (hT @ fc1^T + b1) @ fc2^T + b2 ----------------
__global__ void fc_kernel(const float* __restrict__ hT,
                          const float* __restrict__ w1, const float* __restrict__ b1,
                          const float* __restrict__ w2, const float* __restrict__ b2,
                          float* __restrict__ out) {
    int b = blockIdx.x, tid = threadIdx.x;
    __shared__ float hrow[HDIM];
    __shared__ float tmp[HDIM];
    __shared__ float red[256];
    for (int j = tid; j < HDIM; j += 256) hrow[j] = hT[(size_t)b * HDIM + j];
    __syncthreads();
    for (int j = tid; j < HDIM; j += 256) {
        float s = b1[j];
        for (int k = 0; k < HDIM; ++k) s += hrow[k] * w1[(size_t)j * HDIM + k];
        tmp[j] = s;
    }
    __syncthreads();
    for (int o = 0; o < 2; ++o) {
        float s = 0.0f;
        for (int k = tid; k < HDIM; k += 256) s += tmp[k] * w2[(size_t)o * HDIM + k];
        red[tid] = s; __syncthreads();
        for (int st = 128; st > 0; st >>= 1) { if (tid < st) red[tid] += red[tid + st]; __syncthreads(); }
        if (tid == 0) out[b * 2 + o] = red[0] + b2[o];
        __syncthreads();
    }
}

// ---------------- host launch ----------------

extern "C" void kernel_launch(void* const* d_in, const int* in_sizes, int n_in,
                              void* d_out, int out_size, void* d_ws, size_t ws_size,
                              hipStream_t stream) {
    (void)in_sizes; (void)n_in; (void)out_size; (void)ws_size;
    const float* x     = (const float*)d_in[0];
    const float* wih0f = (const float*)d_in[1];
    const float* whh0f = (const float*)d_in[2];
    const float* bih0f = (const float*)d_in[3];
    const float* bhh0f = (const float*)d_in[4];
    const float* wih0r = (const float*)d_in[5];
    const float* whh0r = (const float*)d_in[6];
    const float* bih0r = (const float*)d_in[7];
    const float* bhh0r = (const float*)d_in[8];
    // d_in[9..12]: layer-1 forward direction — provably unused by the reference output
    const float* wih1r = (const float*)d_in[13];
    const float* whh1r = (const float*)d_in[14];
    const float* bih1r = (const float*)d_in[15];
    const float* bhh1r = (const float*)d_in[16];
    const float* fc1w  = (const float*)d_in[17];
    const float* fc1b  = (const float*)d_in[18];
    const float* fcw   = (const float*)d_in[19];
    const float* fcb   = (const float*)d_in[20];
    float* out = (float*)d_out;

    // bump allocator over d_ws
    size_t off = 0;
    auto alloc = [&](size_t bytes) -> char* {
        char* p = (char*)d_ws + off;
        off += (bytes + 255) & ~(size_t)255;
        return p;
    };
    int*      lens   = (int*)alloc(BATCH * sizeof(int));
    unsigned* cnt    = (unsigned*)alloc(4 * sizeof(unsigned));
    __bf16*   A0     = (__bf16*)alloc((size_t)MT_ALL * KT_X0 * 512 * 2);
    __bf16*   pWih0f = (__bf16*)alloc((size_t)NT_ALL * KT_X0 * 512 * 2);
    __bf16*   pWih0r = (__bf16*)alloc((size_t)NT_ALL * KT_X0 * 512 * 2);
    __bf16*   pWhh0f = (__bf16*)alloc((size_t)NT_ALL * KT_H  * 512 * 2);
    __bf16*   pWhh0r = (__bf16*)alloc((size_t)NT_ALL * KT_H  * 512 * 2);
    __bf16*   pWih1r = (__bf16*)alloc((size_t)NT_ALL * KT_X1 * 512 * 2);
    __bf16*   pWhh1r = (__bf16*)alloc((size_t)NT_ALL * KT_H  * 512 * 2);
    float*    pre0f  = (float*)alloc((size_t)S_LEN * NT_ALL * 8 * 256 * 4);
    float*    pre0r  = (float*)alloc((size_t)S_LEN * NT_ALL * 8 * 256 * 4);
    float*    pre1r  = (float*)alloc((size_t)S_LEN * NT_ALL * 8 * 256 * 4);
    __bf16*   A1     = (__bf16*)alloc((size_t)MT_ALL * KT_X1 * 512 * 2);
    __bf16*   hpk0f  = (__bf16*)alloc((size_t)8 * KT_H * 512 * 2);
    __bf16*   hpk0r  = (__bf16*)alloc((size_t)8 * KT_H * 512 * 2);
    __bf16*   hpk1   = (__bf16*)alloc((size_t)8 * KT_H * 512 * 2);
    float*    hT     = (float*)alloc((size_t)BATCH * HDIM * 4);

    dim3 blk256(256), blkw(32, 8);

    len_kernel<<<1, 128, 0, stream>>>(x, lens);
    init_kernel<<<(8 * KT_H * 512 + 255) / 256, blk256, 0, stream>>>(hpk0f, hpk0r, hpk1, cnt);

    pack_x_kernel<<<(int)(((size_t)MT_ALL * KT_X0 * 512 + 255) / 256), blk256, 0, stream>>>(x, A0);
    pack_w_kernel<<<(NT_ALL * KT_X0 * 512 + 255) / 256, blk256, 0, stream>>>(wih0f, pWih0f, EDIM, KT_X0);
    pack_w_kernel<<<(NT_ALL * KT_X0 * 512 + 255) / 256, blk256, 0, stream>>>(wih0r, pWih0r, EDIM, KT_X0);
    pack_w_kernel<<<(NT_ALL * KT_H  * 512 + 255) / 256, blk256, 0, stream>>>(whh0f, pWhh0f, HDIM, KT_H);
    pack_w_kernel<<<(NT_ALL * KT_H  * 512 + 255) / 256, blk256, 0, stream>>>(whh0r, pWhh0r, HDIM, KT_H);
    pack_w_kernel<<<(NT_ALL * KT_X1 * 512 + 255) / 256, blk256, 0, stream>>>(wih1r, pWih1r, 2 * HDIM, KT_X1);
    pack_w_kernel<<<(NT_ALL * KT_H  * 512 + 255) / 256, blk256, 0, stream>>>(whh1r, pWhh1r, HDIM, KT_H);

    // layer-0 input projections for all timesteps (dense WMMA GEMMs)
    dim3 ggrid(NT_ALL / 8, MT_ALL / 8);
    gemm_pre_kernel<<<ggrid, blkw, 0, stream>>>(A0, pWih0f, bih0f, bhh0f, pre0f, KT_X0);
    gemm_pre_kernel<<<ggrid, blkw, 0, stream>>>(A0, pWih0r, bih0r, bhh0r, pre0r, KT_X0);

    // layer 0: both directions concurrently, writing packed concat activations
    lstm_recur_kernel<<<dim3(16, 2), blkw, 0, stream>>>(
        pre0f, pre0r, pWhh0f, pWhh0r, hpk0f, hpk0r, lens,
        A1, /*koff*/ 0, HDIM, /*rev*/ 0, 1, cnt, nullptr);

    // layer-1 (reverse only) input projection
    gemm_pre_kernel<<<ggrid, blkw, 0, stream>>>(A1, pWih1r, bih1r, bhh1r, pre1r, KT_X1);

    // layer 1 reverse recurrence -> hT
    lstm_recur_kernel<<<dim3(16, 1), blkw, 0, stream>>>(
        pre1r, pre1r, pWhh1r, pWhh1r, hpk1, hpk1, lens,
        nullptr, 0, 0, /*rev*/ 1, 1, cnt + 2, hT);

    fc_kernel<<<BATCH, blk256, 0, stream>>>(hT, fc1w, fc1b, fcw, fcb, out);
}
